// MultiHeadAttention_9972914062040
// MI455X (gfx1250) — matrix operand order repaired and checked by reading, not compiled, not run on hardware
//
#include <hip/hip_runtime.h>
#include <hip/hip_bf16.h>

// Problem constants (match reference)
#define B_ 4
#define L_ 1024
#define H_ 16
#define D_ 64
#define C_ 1024
#define M_ (B_ * L_)     // 4096 tokens
#define EPS_ 1e-5f
#define THETA_ 50000.0f

typedef unsigned short u16;
typedef unsigned int   u32;
typedef __attribute__((ext_vector_type(16))) __bf16 bf16x16;
typedef __attribute__((ext_vector_type(8)))  float  v8f;

// ---------------- helpers ----------------

__device__ __forceinline__ u16 f2bf(float f) {          // round-to-nearest-even bf16
  u32 u = __float_as_uint(f);
  u32 r = u + 0x7FFFu + ((u >> 16) & 1u);
  return (u16)(r >> 16);
}

__device__ __forceinline__ bf16x16 frag_from(uint4 lo, uint4 hi) {
  union { bf16x16 v; uint4 q[2]; } un;
  un.q[0] = lo; un.q[1] = hi;
  return un.v;
}

// A fragment (16x32 bf16, MxK). Caller passes &row[k0 + kb_a] with
// kb_a = (lane<16 ? 0 : 8). Elements 0..7 = K kb..kb+7, 8..15 = K kb+16..kb+23.
__device__ __forceinline__ bf16x16 load_a(const u16* p) {
  const uint4* q = (const uint4*)p;
  return frag_from(q[0], q[2]);          // +0 and +16 elements
}

// B fragment (32x16 bf16, KxN). Caller passes &col_row[k0 + kb_b] with
// kb_b = (lane<16 ? 0 : 8). Elements 0..7 = K kb..kb+7, 8..15 = K kb+16..kb+23.
__device__ __forceinline__ bf16x16 load_b(const u16* p) {
  const uint4* q = (const uint4*)p;
  return frag_from(q[0], q[2]);          // +0 and +16 elements
}

__device__ __forceinline__ v8f wmma_bf16(bf16x16 a, bf16x16 b, v8f c) {
  return __builtin_amdgcn_wmma_f32_16x16x32_bf16(false, a, false, b, (short)0, c,
                                                 false, false);
}

// reductions within each 16-lane half of the wave32 (row lives across 16 lanes)
__device__ __forceinline__ float hsum16(float x) {
  x += __shfl_xor(x, 1, 16);
  x += __shfl_xor(x, 2, 16);
  x += __shfl_xor(x, 4, 16);
  x += __shfl_xor(x, 8, 16);
  return x;
}
__device__ __forceinline__ float hmax16(float x) {
  x = fmaxf(x, __shfl_xor(x, 1, 16));
  x = fmaxf(x, __shfl_xor(x, 2, 16));
  x = fmaxf(x, __shfl_xor(x, 4, 16));
  x = fmaxf(x, __shfl_xor(x, 8, 16));
  return x;
}

// ---------------- f32 -> bf16 convert ----------------

__global__ void k_cvt(const float* __restrict__ src, u16* __restrict__ dst, int n) {
  int i = blockIdx.x * blockDim.x + threadIdx.x;
  if (i < n) dst[i] = f2bf(src[i]);
}

// ---------------- QKV projection GEMM + fused LN + RoPE ----------------
// X [M_,C_] bf16, W [C_,C_] bf16 row-major [o][c]  (y = x @ W^T)
// mode 0/1: LayerNorm(D)+RoPE, store bf16 [B,H,L,D]
// mode 2  : store bf16 transposed [B,H,D,L]
// Block 256 thr = 8 waves; wave -> 16 rows x 64 cols (one full head).
__global__ __launch_bounds__(256) void k_gemm_qkv(
    const u16* __restrict__ X, const u16* __restrict__ W,
    u16* __restrict__ out,
    const float* __restrict__ lnw, const float* __restrict__ lnb, int mode) {
  const int lane   = threadIdx.x & 31;
  const int wave   = threadIdx.x >> 5;
  const int lane15 = lane & 15;
  const int lanehi = lane >> 4;
  const int row0   = blockIdx.x * 128 + wave * 16;   // token rows
  const int n0     = blockIdx.y * 64;                // channel strip = head
  const int h      = n0 / D_;

  v8f acc[4];
#pragma unroll
  for (int j = 0; j < 4; ++j) { v8f z = {}; acc[j] = z; }

  const int kb_a = lanehi ? 8 : 0;
  const int kb_b = lanehi ? 8 : 0;
  const u16* arow = X + (size_t)(row0 + lane15) * C_ + kb_a;
  const u16* wrow = W + (size_t)(n0 + lane15) * C_ + kb_b;

  for (int kk = 0; kk < C_; kk += 32) {
    bf16x16 a = load_a(arow + kk);
#pragma unroll
    for (int j = 0; j < 4; ++j) {
      bf16x16 b = load_b(wrow + (size_t)j * 16 * C_ + kk);
      acc[j] = wmma_bf16(a, b, acc[j]);
    }
  }

  const int b_idx = row0 / L_;
  const int s0    = row0 % L_;

  if (mode == 2) {
    // V: store transposed [B,H,D,L]; 8 rows per lane are contiguous in s.
#pragma unroll
    for (int j = 0; j < 4; ++j) {
      const int d = j * 16 + lane15;
      alignas(16) u16 pk[8];
#pragma unroll
      for (int r = 0; r < 8; ++r) pk[r] = f2bf(acc[j][r]);
      u16* dst = out + ((size_t)(b_idx * H_ + h) * D_ + d) * L_ + s0 + 8 * lanehi;
      *(uint4*)dst = *(const uint4*)pk;
    }
    return;
  }

  // Q/K: per-head LayerNorm over D=64 + rotary embedding.
  float wv[4], bv[4];
#pragma unroll
  for (int j = 0; j < 4; ++j) {
    int d = j * 16 + lane15;
    wv[j] = lnw[d];
    bv[j] = lnb[d];
  }
  const float l2t = __log2f(THETA_);
  float invf[2];
  invf[0] = exp2f(-((float)lane15 / 32.0f) * l2t);          // d%32 = lane15
  invf[1] = exp2f(-((float)(16 + lane15) / 32.0f) * l2t);   // d%32 = 16+lane15

#pragma unroll
  for (int r = 0; r < 8; ++r) {
    const int m   = r + 8 * lanehi;          // C/D layout: low lanes M=r, high M=8+r
    const int pos = s0 + m;
    float x0 = acc[0][r], x1 = acc[1][r], x2 = acc[2][r], x3 = acc[3][r];
    float mu = hsum16(x0 + x1 + x2 + x3) * (1.0f / 64.0f);
    float d0 = x0 - mu, d1 = x1 - mu, d2 = x2 - mu, d3 = x3 - mu;
    float var  = hsum16(d0 * d0 + d1 * d1 + d2 * d2 + d3 * d3) * (1.0f / 64.0f);
    float rstd = rsqrtf(var + EPS_);
    float nv[4];
    nv[0] = d0 * rstd * wv[0] + bv[0];
    nv[1] = d1 * rstd * wv[1] + bv[1];
    nv[2] = d2 * rstd * wv[2] + bv[2];
    nv[3] = d3 * rstd * wv[3] + bv[3];
    // RoPE: rotate_half partner d±32 lives in tile j±2, same lane, same r.
    u16* dst = out + ((size_t)(b_idx * H_ + h) * L_ + pos) * D_;
#pragma unroll
    for (int j = 0; j < 4; ++j) {
      float ang = (float)pos * invf[j & 1];
      float sn, cs;
      __sincosf(ang, &sn, &cs);
      float rot = (j < 2) ? -nv[j + 2] : nv[j - 2];
      dst[j * 16 + lane15] = f2bf(nv[j] * cs + rot * sn);
    }
  }
}

// ---------------- fused flash attention ----------------
// Q,K bf16 [B,H,L,D]; V bf16 transposed [B,H,D,L]; out bf16 [B,L,C].
// Block 256 thr = 8 waves; wave -> 16 query rows, loops KV in 64-key blocks.
__global__ __launch_bounds__(256) void k_attn(
    const u16* __restrict__ Q, const u16* __restrict__ Kq,
    const u16* __restrict__ Vt, u16* __restrict__ Ao) {
  __shared__ __align__(16) u16 pls[8][16 * 64];    // P tile per wave (bf16)

  const int lane   = threadIdx.x & 31;
  const int wave   = threadIdx.x >> 5;
  const int lane15 = lane & 15;
  const int lanehi = lane >> 4;
  const int bh     = blockIdx.x;                   // 0..63
  const int b_idx  = bh / H_;
  const int h      = bh % H_;
  const int q0     = blockIdx.y * 128 + wave * 16;

  const u16* Qh = Q  + (size_t)bh * L_ * D_;
  const u16* Kh = Kq + (size_t)bh * L_ * D_;
  const u16* Vh = Vt + (size_t)bh * D_ * L_;

  const int kb_a = lanehi ? 8 : 0;
  const int kb_b = lanehi ? 8 : 0;

  // Q A-fragments (K-dim = d, 2 steps of 32) — loaded once.
  const u16* qrow = Qh + (size_t)(q0 + lane15) * D_ + kb_a;
  bf16x16 aQ0 = load_a(qrow);
  bf16x16 aQ1 = load_a(qrow + 32);

  v8f oAcc[4];
#pragma unroll
  for (int j = 0; j < 4; ++j) { v8f z = {}; oAcc[j] = z; }
  float mrow[8], lrow[8];
#pragma unroll
  for (int r = 0; r < 8; ++r) { mrow[r] = -3.0e38f; lrow[r] = 0.0f; }

  u16* myP = &pls[wave][0];
  const float sc = 0.125f;                         // 1/sqrt(64)

  for (int s0 = 0; s0 < L_; s0 += 64) {
    // ---- S = Q K^T (16 x 64) ----
    v8f sAcc[4];
#pragma unroll
    for (int j = 0; j < 4; ++j) { v8f z = {}; sAcc[j] = z; }
#pragma unroll
    for (int j = 0; j < 4; ++j) {
      const u16* krow = Kh + (size_t)(s0 + j * 16 + lane15) * D_ + kb_b;
      sAcc[j] = wmma_bf16(aQ0, load_b(krow),      sAcc[j]);
      sAcc[j] = wmma_bf16(aQ1, load_b(krow + 32), sAcc[j]);
    }
    // ---- online softmax over this 64-key block ----
#pragma unroll
    for (int r = 0; r < 8; ++r) {
      float s0v = sAcc[0][r] * sc, s1v = sAcc[1][r] * sc;
      float s2v = sAcc[2][r] * sc, s3v = sAcc[3][r] * sc;
      float bm    = hmax16(fmaxf(fmaxf(s0v, s1v), fmaxf(s2v, s3v)));
      float mnew  = fmaxf(mrow[r], bm);
      float alpha = __expf(mrow[r] - mnew);
      float p0 = __expf(s0v - mnew), p1 = __expf(s1v - mnew);
      float p2 = __expf(s2v - mnew), p3 = __expf(s3v - mnew);
      lrow[r] = lrow[r] * alpha + hsum16(p0 + p1 + p2 + p3);
      mrow[r] = mnew;
#pragma unroll
      for (int j = 0; j < 4; ++j) oAcc[j][r] = oAcc[j][r] * alpha;
      // C-layout -> LDS (row = r + 8*lanehi, col = j*16 + lane15)
      u16* prow = myP + (size_t)(r + 8 * lanehi) * 64 + lane15;
      prow[0] = f2bf(p0); prow[16] = f2bf(p1); prow[32] = f2bf(p2); prow[48] = f2bf(p3);
    }
    asm volatile("s_wait_dscnt 0" ::: "memory");   // LDS stores visible (same wave)
    // ---- reload P in A-layout, O += P V ----
    const u16* pr = myP + (size_t)lane15 * 64 + kb_a;
    bf16x16 aP0 = load_a(pr);
    bf16x16 aP1 = load_a(pr + 32);
#pragma unroll
    for (int j = 0; j < 4; ++j) {
      const u16* vrow = Vh + (size_t)(j * 16 + lane15) * L_ + s0 + kb_b;
      oAcc[j] = wmma_bf16(aP0, load_b(vrow),      oAcc[j]);
      oAcc[j] = wmma_bf16(aP1, load_b(vrow + 32), oAcc[j]);
    }
  }

  // ---- normalize and store bf16 [B,L,C] ----
#pragma unroll
  for (int r = 0; r < 8; ++r) {
    float inv = 1.0f / lrow[r];
    int qr = q0 + r + 8 * lanehi;
    u16* dst = Ao + (size_t)(b_idx * L_ + qr) * C_ + h * D_ + lane15;
#pragma unroll
    for (int j = 0; j < 4; ++j) dst[j * 16] = f2bf(oAcc[j][r] * inv);
  }
}

// ---------------- output projection GEMM + bias (f32 out) ----------------
__global__ __launch_bounds__(256) void k_gemm_out(
    const u16* __restrict__ A, const u16* __restrict__ W,
    const float* __restrict__ bo, float* __restrict__ out) {
  const int lane   = threadIdx.x & 31;
  const int wave   = threadIdx.x >> 5;
  const int lane15 = lane & 15;
  const int lanehi = lane >> 4;
  const int row0   = blockIdx.x * 128 + wave * 16;
  const int n0     = blockIdx.y * 64;

  v8f acc[4];
#pragma unroll
  for (int j = 0; j < 4; ++j) { v8f z = {}; acc[j] = z; }

  const int kb_a = lanehi ? 8 : 0;
  const int kb_b = lanehi ? 8 : 0;
  const u16* arow = A + (size_t)(row0 + lane15) * C_ + kb_a;
  const u16* wrow = W + (size_t)(n0 + lane15) * C_ + kb_b;

  for (int kk = 0; kk < C_; kk += 32) {
    bf16x16 a = load_a(arow + kk);
#pragma unroll
    for (int j = 0; j < 4; ++j) {
      bf16x16 b = load_b(wrow + (size_t)j * 16 * C_ + kk);
      acc[j] = wmma_bf16(a, b, acc[j]);
    }
  }

  float bias[4];
#pragma unroll
  for (int j = 0; j < 4; ++j) bias[j] = bo[n0 + j * 16 + lane15];
#pragma unroll
  for (int r = 0; r < 8; ++r) {
    int row = row0 + r + 8 * lanehi;
    float* dst = out + (size_t)row * C_ + n0 + lane15;
#pragma unroll
    for (int j = 0; j < 4; ++j) dst[j * 16] = acc[j][r] + bias[j];
  }
}

// ---------------- launch ----------------

extern "C" void kernel_launch(void* const* d_in, const int* in_sizes, int n_in,
                              void* d_out, int out_size, void* d_ws, size_t ws_size,
                              hipStream_t stream) {
  (void)in_sizes; (void)n_in; (void)out_size; (void)ws_size;
  const float* x   = (const float*)d_in[0];
  const float* Wq  = (const float*)d_in[1];
  const float* Wk  = (const float*)d_in[2];
  const float* Wv  = (const float*)d_in[3];
  const float* Wo  = (const float*)d_in[4];
  const float* bo  = (const float*)d_in[5];
  const float* qnw = (const float*)d_in[6];
  const float* qnb = (const float*)d_in[7];
  const float* knw = (const float*)d_in[8];
  const float* knb = (const float*)d_in[9];

  char* ws = (char*)d_ws;
  size_t off = 0;
  auto alloc = [&](size_t bytes) -> char* {
    char* p = ws + off;
    off += (bytes + 255) & ~(size_t)255;
    return p;
  };
  u16* Xb  = (u16*)alloc((size_t)M_ * C_ * 2);        // 8 MB  x bf16
  u16* Wqb = (u16*)alloc((size_t)C_ * C_ * 2);        // 2 MB
  u16* Wkb = (u16*)alloc((size_t)C_ * C_ * 2);
  u16* Wvb = (u16*)alloc((size_t)C_ * C_ * 2);
  u16* Wob = (u16*)alloc((size_t)C_ * C_ * 2);
  u16* Qb  = (u16*)alloc((size_t)M_ * C_ * 2);        // [B,H,L,D]
  u16* Kb  = (u16*)alloc((size_t)M_ * C_ * 2);        // [B,H,L,D]
  u16* Vb  = (u16*)alloc((size_t)M_ * C_ * 2);        // [B,H,D,L]
  u16* Ab  = (u16*)alloc((size_t)M_ * C_ * 2);        // attn out [B,L,C]

  const int NX = M_ * C_, NW = C_ * C_;
  k_cvt<<<(NX + 255) / 256, 256, 0, stream>>>(x,  Xb,  NX);
  k_cvt<<<(NW + 255) / 256, 256, 0, stream>>>(Wq, Wqb, NW);
  k_cvt<<<(NW + 255) / 256, 256, 0, stream>>>(Wk, Wkb, NW);
  k_cvt<<<(NW + 255) / 256, 256, 0, stream>>>(Wv, Wvb, NW);
  k_cvt<<<(NW + 255) / 256, 256, 0, stream>>>(Wo, Wob, NW);

  dim3 gg(M_ / 128, C_ / 64);                         // 32 x 16
  k_gemm_qkv<<<gg, 256, 0, stream>>>(Xb, Wqb, Qb, qnw, qnb, 0);
  k_gemm_qkv<<<gg, 256, 0, stream>>>(Xb, Wkb, Kb, knw, knb, 1);
  k_gemm_qkv<<<gg, 256, 0, stream>>>(Xb, Wvb, Vb, qnw, qnb, 2);

  k_attn<<<dim3(B_ * H_, L_ / 128), 256, 0, stream>>>(Qb, Kb, Vb, Ab);

  k_gemm_out<<<gg, 256, 0, stream>>>(Ab, Wob, bo, (float*)d_out);
}
